// EdgePredictor_5858335392468
// MI455X (gfx1250) — matrix operand order repaired and checked by reading, not compiled, not run on hardware
//
#include <hip/hip_runtime.h>
#include <hip/hip_bf16.h>

typedef __attribute__((ext_vector_type(16))) __bf16 v16bf;
typedef __attribute__((ext_vector_type(8)))  __bf16 v8bf;
typedef __attribute__((ext_vector_type(4)))  __bf16 v4bf;
typedef __attribute__((ext_vector_type(8)))  float  v8f;

static constexpr int N   = 8192;
static constexpr int K   = 128;
static constexpr int TI  = 64;        // output rows per block
static constexpr int TJ  = 32;        // output cols per block
static constexpr int LDH = K + 8;     // padded LDS row stride (halves) -> spreads banks

// scores[i][j] = <h[i], h[j]>, diagonal zeroed.
// Split-bf16 fp32 emulation: x = hi + lo; A*B ~= Ahi*Bhi + Ahi*Blo + Alo*Bhi (fp32 accum).
__global__ __launch_bounds__(256) void pairwise_scores_wmma(const float* __restrict__ h,
                                                            float* __restrict__ out) {
    __shared__ __bf16 Ahi[TI * LDH];
    __shared__ __bf16 Alo[TI * LDH];
    __shared__ __bf16 Bhi[TJ * LDH];
    __shared__ __bf16 Blo[TJ * LDH];

    const int tid = threadIdx.x;
    const int i0  = blockIdx.y * TI;   // global row base
    const int j0  = blockIdx.x * TJ;   // global col base

    const float4* hv = reinterpret_cast<const float4*>(h);  // row stride = K/4 float4

    // ---- Stage A rows (hi/lo bf16 limbs) ----
    for (int idx = tid; idx < TI * (K / 4); idx += 256) {
        const int r  = idx >> 5;        // row in tile
        const int c4 = idx & 31;        // float4 column
        float4 x = hv[(size_t)(i0 + r) * (K / 4) + c4];
        float xs[4] = {x.x, x.y, x.z, x.w};
        v4bf hiv, lov;
#pragma unroll
        for (int e = 0; e < 4; ++e) {
            __bf16 hb = (__bf16)xs[e];
            hiv[e] = hb;
            lov[e] = (__bf16)(xs[e] - (float)hb);
        }
        *reinterpret_cast<v4bf*>(&Ahi[r * LDH + c4 * 4]) = hiv;
        *reinterpret_cast<v4bf*>(&Alo[r * LDH + c4 * 4]) = lov;
    }
    // ---- Stage B rows (hi/lo bf16 limbs) ----
    for (int idx = tid; idx < TJ * (K / 4); idx += 256) {
        const int r  = idx >> 5;
        const int c4 = idx & 31;
        float4 x = hv[(size_t)(j0 + r) * (K / 4) + c4];
        float xs[4] = {x.x, x.y, x.z, x.w};
        v4bf hiv, lov;
#pragma unroll
        for (int e = 0; e < 4; ++e) {
            __bf16 hb = (__bf16)xs[e];
            hiv[e] = hb;
            lov[e] = (__bf16)(xs[e] - (float)hb);
        }
        *reinterpret_cast<v4bf*>(&Bhi[r * LDH + c4 * 4]) = hiv;
        *reinterpret_cast<v4bf*>(&Blo[r * LDH + c4 * 4]) = lov;
    }
    __syncthreads();

    // ---- wave32 subtile assignment: 8 waves -> 4x2 grid of 16x16 subtiles ----
    const int lane = tid & 31;
    const int wave = tid >> 5;
    const int lrow = lane & 15;   // A row index / B col index within subtile
    const int lhi  = lane >> 4;   // lane half select
    const int sr   = wave >> 1;   // subtile row 0..3
    const int sc   = wave & 1;    // subtile col 0..1

    const __bf16* aHiRow = &Ahi[(sr * 16 + lrow) * LDH];
    const __bf16* aLoRow = &Alo[(sr * 16 + lrow) * LDH];
    const __bf16* bHiRow = &Bhi[(sc * 16 + lrow) * LDH];
    const __bf16* bLoRow = &Blo[(sc * 16 + lrow) * LDH];

    v8f acc = {};
#pragma unroll
    for (int k0 = 0; k0 < K; k0 += 32) {
        // A 16x32 bf16 fragment: per lane K = {k0+8*lhi..+7} U {k0+16+8*lhi..+7}
        v8bf ah0 = *reinterpret_cast<const v8bf*>(aHiRow + k0 + 8 * lhi);
        v8bf ah1 = *reinterpret_cast<const v8bf*>(aHiRow + k0 + 16 + 8 * lhi);
        v8bf al0 = *reinterpret_cast<const v8bf*>(aLoRow + k0 + 8 * lhi);
        v8bf al1 = *reinterpret_cast<const v8bf*>(aLoRow + k0 + 16 + 8 * lhi);
        v16bf ahi = __builtin_shufflevector(ah0, ah1, 0,1,2,3,4,5,6,7,8,9,10,11,12,13,14,15);
        v16bf alo = __builtin_shufflevector(al0, al1, 0,1,2,3,4,5,6,7,8,9,10,11,12,13,14,15);
        // B 32x16 bf16 fragment: per lane K = {k0+8*lhi..+7} U {k0+16+8*lhi..+7}; N = lrow
        v8bf bh0 = *reinterpret_cast<const v8bf*>(bHiRow + k0 + 8 * lhi);
        v8bf bh1 = *reinterpret_cast<const v8bf*>(bHiRow + k0 + 16 + 8 * lhi);
        v8bf bl0 = *reinterpret_cast<const v8bf*>(bLoRow + k0 + 8 * lhi);
        v8bf bl1 = *reinterpret_cast<const v8bf*>(bLoRow + k0 + 16 + 8 * lhi);
        v16bf bhi = __builtin_shufflevector(bh0, bh1, 0,1,2,3,4,5,6,7,8,9,10,11,12,13,14,15);
        v16bf blo = __builtin_shufflevector(bl0, bl1, 0,1,2,3,4,5,6,7,8,9,10,11,12,13,14,15);

        acc = __builtin_amdgcn_wmma_f32_16x16x32_bf16(false, ahi, false, bhi,
                                                      (short)0, acc, false, false);
        acc = __builtin_amdgcn_wmma_f32_16x16x32_bf16(false, ahi, false, blo,
                                                      (short)0, acc, false, false);
        acc = __builtin_amdgcn_wmma_f32_16x16x32_bf16(false, alo, false, bhi,
                                                      (short)0, acc, false, false);
    }

    // ---- store: C/D layout -> vgpr v holds row (v + 8*lhi), col = lrow ----
    const int gj     = j0 + sc * 16 + lrow;
    const int giBase = i0 + sr * 16 + lhi * 8;
#pragma unroll
    for (int v = 0; v < 8; ++v) {
        const int gi = giBase + v;
        out[(size_t)gi * N + gj] = (gi == gj) ? 0.0f : acc[v];
    }
}

extern "C" void kernel_launch(void* const* d_in, const int* in_sizes, int n_in,
                              void* d_out, int out_size, void* d_ws, size_t ws_size,
                              hipStream_t stream) {
    (void)in_sizes; (void)n_in; (void)out_size; (void)d_ws; (void)ws_size;
    const float* h = (const float*)d_in[0];
    float* out = (float*)d_out;
    dim3 grid(N / TJ, N / TI);   // (256, 128)
    pairwise_scores_wmma<<<grid, dim3(256), 0, stream>>>(h, out);
}
